// PaiConv_76381698392131
// MI455X (gfx1250) — compile-verified, matrix operand order checked by reading, not run on hardware
//
#include <hip/hip_runtime.h>
#include <hip/hip_bf16.h>

typedef __attribute__((ext_vector_type(16))) _Float16 v16h;
typedef __attribute__((ext_vector_type(8)))  float    v8f;

#define NODES 16384
#define BATCH 16
#define CIN   128
#define KNB   9          // neighbours
#define FDIM  1152       // KNB * CIN
#define COUT  128
#define LDA   1160       // padded f16 row stride for conv A panel (2320B, 16B-mult, bank-spread)
#define LDX   136        // padded f16 row stride for skip A panel (272B)

union Frag16 { v16h h; uint4 q[2]; };

__device__ __forceinline__ float elu_f(float v) {
    return v > 0.0f ? v : (__expf(v) - 1.0f);
}

// ---------------------------------------------------------------------------
// Pre-pass: convert W_conv (128x1152) and W_skip (128x128) f32 -> f16 in d_ws.
// Row-major layout kept: a WMMA B-fragment lane (= one output channel column)
// then reads two contiguous 16B chunks per k-step.
// ---------------------------------------------------------------------------
__global__ void convert_weights_kernel(const float* __restrict__ Wc,
                                       const float* __restrict__ Ws,
                                       _Float16* __restrict__ Wc16,
                                       _Float16* __restrict__ Ws16) {
    int i = blockIdx.x * 256 + threadIdx.x;
    if (i < COUT * FDIM) Wc16[i] = (_Float16)Wc[i];
    if (i < COUT * CIN)  Ws16[i] = (_Float16)Ws[i];
}

// ---------------------------------------------------------------------------
// Main kernel: one block (4 waves / 128 threads) per node n.
//   Stage 1 (VALU fp32): gather 9 neighbours, 9x9 P-mix, ELU, f16 A panel->LDS
//   Stage 2 (WMMA f16):  [16 x 1152] x [1152 x 128] conv GEMM (36 k-steps)
//                        + [16 x 128] x [128 x 128] skip GEMM (4 k-steps)
//   Epilogue: +b_conv, ELU, +skip +b_skip, f32 store.
// ---------------------------------------------------------------------------
__global__ void __launch_bounds__(128)
paiconv_wmma_kernel(const float* __restrict__ x,
                    const int*   __restrict__ idx,
                    const float* __restrict__ P,
                    const _Float16* __restrict__ Wc16,
                    const float* __restrict__ bc,
                    const _Float16* __restrict__ Ws16,
                    const float* __restrict__ bs,
                    float* __restrict__ out) {
    __shared__ __align__(16) _Float16 Apan[BATCH * LDA];  // 36.25 KB
    __shared__ __align__(16) _Float16 Axs [BATCH * LDX];  //  4.25 KB
    __shared__ float Psh[KNB * KNB];
    __shared__ int   nbr_idx[KNB];

    const int n = blockIdx.x;
    const int t = threadIdx.x;            // 0..127

    if (t < KNB * KNB) Psh[t] = P[(size_t)n * (KNB * KNB) + t];
    if (t < KNB)       nbr_idx[t] = idx[n * KNB + t];
    __syncthreads();

    // ---------------- Stage 1: gather + P-mix + ELU -> f16 LDS panel -------
    {
        const int c = t;                  // one channel per thread
        for (int b = 0; b < BATCH; ++b) {
            float nbr[KNB];
#pragma unroll
            for (int j = 0; j < KNB; ++j)
                nbr[j] = x[((size_t)(b * NODES + nbr_idx[j])) * CIN + c];
#pragma unroll
            for (int k = 0; k < KNB; ++k) {
                float s = 0.0f;
#pragma unroll
                for (int j = 0; j < KNB; ++j)
                    s = fmaf(Psh[k * KNB + j], nbr[j], s);
                Apan[b * LDA + k * CIN + c] = (_Float16)elu_f(s);
            }
            Axs[b * LDX + c] = (_Float16)x[((size_t)(b * NODES + n)) * CIN + c];
        }
    }
    __syncthreads();

    // ---------------- Stage 2: WMMA GEMMs ----------------------------------
    const int lane = t & 31;
    const int wave = t >> 5;
    const int col  = lane & 15;           // N index within tile / A row M
    const int hi   = lane >> 4;           // K-half select per ISA layout
    const int nt0  = wave * 2;            // two output-channel tiles per wave
    const int nt1  = wave * 2 + 1;
    const int o0   = nt0 * 16 + col;
    const int o1   = nt1 * 16 + col;

    v8f accC0 = {}, accC1 = {};
#pragma unroll 4
    for (int kt = 0; kt < FDIM / 32; ++kt) {
        const int ka = kt * 32 + hi * 8;
        Frag16 a, b0, b1;
        a.q[0] = *(const uint4*)&Apan[col * LDA + ka];
        a.q[1] = *(const uint4*)&Apan[col * LDA + ka + 16];
        const _Float16* w0 = Wc16 + (size_t)o0 * FDIM + ka;
        const _Float16* w1 = Wc16 + (size_t)o1 * FDIM + ka;
        b0.q[0] = *(const uint4*)(w0);      b0.q[1] = *(const uint4*)(w0 + 16);
        b1.q[0] = *(const uint4*)(w1);      b1.q[1] = *(const uint4*)(w1 + 16);
        accC0 = __builtin_amdgcn_wmma_f32_16x16x32_f16(false, a.h, false, b0.h,
                                                       (short)0, accC0, false, false);
        accC1 = __builtin_amdgcn_wmma_f32_16x16x32_f16(false, a.h, false, b1.h,
                                                       (short)0, accC1, false, false);
    }

    v8f accS0 = {}, accS1 = {};
#pragma unroll
    for (int kt = 0; kt < CIN / 32; ++kt) {
        const int ka = kt * 32 + hi * 8;
        Frag16 a, b0, b1;
        a.q[0] = *(const uint4*)&Axs[col * LDX + ka];
        a.q[1] = *(const uint4*)&Axs[col * LDX + ka + 16];
        const _Float16* w0 = Ws16 + (size_t)o0 * CIN + ka;
        const _Float16* w1 = Ws16 + (size_t)o1 * CIN + ka;
        b0.q[0] = *(const uint4*)(w0);      b0.q[1] = *(const uint4*)(w0 + 16);
        b1.q[0] = *(const uint4*)(w1);      b1.q[1] = *(const uint4*)(w1 + 16);
        accS0 = __builtin_amdgcn_wmma_f32_16x16x32_f16(false, a.h, false, b0.h,
                                                       (short)0, accS0, false, false);
        accS1 = __builtin_amdgcn_wmma_f32_16x16x32_f16(false, a.h, false, b1.h,
                                                       (short)0, accS1, false, false);
    }

    // ---------------- Epilogue: bias + ELU + skip, f32 store ---------------
    const float bc0 = bc[o0], bc1 = bc[o1];
    const float bs0 = bs[o0], bs1 = bs[o1];
#pragma unroll
    for (int r = 0; r < 8; ++r) {
        const int brow = r + hi * 8;      // D layout: VGPR r holds M=r (lanes<16) / r+8
        float z0 = elu_f(accC0[r] + bc0) + accS0[r] + bs0;
        float z1 = elu_f(accC1[r] + bc1) + accS1[r] + bs1;
        const size_t base = ((size_t)(brow * NODES + n)) * COUT;
        out[base + o0] = z0;
        out[base + o1] = z1;
    }
}

extern "C" void kernel_launch(void* const* d_in, const int* in_sizes, int n_in,
                              void* d_out, int out_size, void* d_ws, size_t ws_size,
                              hipStream_t stream) {
    const float* x      = (const float*)d_in[0];
    const int*   idx    = (const int*)  d_in[1];
    const float* P      = (const float*)d_in[2];
    const float* W_conv = (const float*)d_in[3];
    const float* b_conv = (const float*)d_in[4];
    const float* W_skip = (const float*)d_in[5];
    const float* b_skip = (const float*)d_in[6];
    float* out = (float*)d_out;

    _Float16* Wc16 = (_Float16*)d_ws;                 // 128*1152 f16 = 288 KB
    _Float16* Ws16 = Wc16 + (size_t)COUT * FDIM;      // 128*128  f16 =  32 KB

    convert_weights_kernel<<<(COUT * FDIM + 255) / 256, 256, 0, stream>>>(
        W_conv, W_skip, Wc16, Ws16);

    paiconv_wmma_kernel<<<NODES, 128, 0, stream>>>(
        x, idx, P, Wc16, b_conv, Ws16, b_skip, out);
}
